// StructNConv2D_e_with_d_35390530519219
// MI455X (gfx1250) — matrix operand order repaired and checked by reading, not compiled, not run on hardware
//
#include <hip/hip_runtime.h>
#include <stdint.h>

#define EPSF 1e-20f

#define Bn 4
#define Cn 32
#define Hn 192
#define Wn 192
#define Hp 194
#define Wp 194
#define C4 128
#define HWn (Hn*Wn)                 // 36864
#define Pn  (Bn*Hn*Wn)              // 147456 pixels (GEMM M)
#define STATS_ELEMS (Bn*Cn*Hp*Wp)   // 4,817,408

typedef __attribute__((ext_vector_type(16))) __bf16 v16bf;
typedef __attribute__((ext_vector_type(8)))  __bf16 v8bf;
typedef __attribute__((ext_vector_type(8)))  float  v8f;

__device__ __forceinline__ unsigned short f2bf(float f) {
  unsigned u = __float_as_uint(f);
  u += 0x7fffu + ((u >> 16) & 1u);   // round-to-nearest-even
  return (unsigned short)(u >> 16);
}

// ---------------------------------------------------------------- prep:
// channel_weight -> bf16 (GEMM B operand), plus sum(cw), sum(sw)
__global__ void k_prep(const float* __restrict__ cw, const float* __restrict__ sw,
                       unsigned short* __restrict__ cwb, float* __restrict__ sums) {
  __shared__ float red[256];
  const int t = threadIdx.x;
  float s1 = 0.f;
  for (int i = t; i < C4*C4; i += 256) { float v = cw[i]; s1 += v; cwb[i] = f2bf(v); }
  float s2 = 0.f;
  for (int i = t; i < C4*25; i += 256) s2 += sw[i];
  red[t] = s1; __syncthreads();
  for (int s = 128; s > 0; s >>= 1) { if (t < s) red[t] += red[t+s]; __syncthreads(); }
  if (t == 0) sums[0] = red[0];
  __syncthreads();
  red[t] = s2; __syncthreads();
  for (int s = 128; s > 0; s >>= 1) { if (t < s) red[t] += red[t+s]; __syncthreads(); }
  if (t == 0) sums[1] = red[0];
}

// ---------------------------------------------------------------- stage 0:
// per padded pixel: 3x3 argmax of d*cd (-> max stats) and cd/(d+eps) (-> min stats)
__global__ void k_stats(const float* __restrict__ d, const float* __restrict__ cd,
                        float* __restrict__ dmin, float* __restrict__ dmax,
                        float* __restrict__ cdmin, float* __restrict__ cdmax) {
  const int idx = blockIdx.x * blockDim.x + threadIdx.x;
  if (idx >= STATS_ELEMS) return;
  const int px = idx % Wp;
  const int py = (idx / Wp) % Hp;
  const int bc = idx / (Wp * Hp);
  const float* __restrict__ dp  = d  + (size_t)bc * HWn;
  const float* __restrict__ cdp = cd + (size_t)bc * HWn;
  float bk1 = -1.f, bk2 = -1.f;          // keys are >= 0; first-wins via strict >
  float vdmax = 0.f, vcdmax = 0.f, vdmin = 0.f, vcdmin = 0.f;
  #pragma unroll
  for (int ddy = -1; ddy <= 1; ++ddy) {
    #pragma unroll
    for (int ddx = -1; ddx <= 1; ++ddx) {
      const int yy = py + ddy, xx = px + ddx;
      if (yy < 0 || yy >= Hp || xx < 0 || xx >= Wp) continue;  // -inf key
      const int oy = yy - 1, ox = xx - 1;
      const bool ins = (oy >= 0) && (oy < Hn) && (ox >= 0) && (ox < Wn);
      const float dv  = ins ? dp[oy*Wn + ox]  : 0.f;   // pad ring = 0
      const float cdv = ins ? cdp[oy*Wn + ox] : 0.f;
      const float k1 = dv * cdv;
      const float k2 = cdv / (dv + EPSF);
      if (k1 > bk1) { bk1 = k1; vdmax = dv; vcdmax = cdv; }
      if (k2 > bk2) { bk2 = k2; vdmin = dv; vcdmin = cdv; }
    }
  }
  dmin[idx] = vdmin; dmax[idx] = vdmax; cdmin[idx] = vcdmin; cdmax[idx] = vcdmax;
}

// ---------------------------------------------------------------- stage 1:
// pair logic + blend; emit X = ce2*e2 and Y = ce2 as bf16 in [pixel][128] layout
__global__ void k_fuse(const float* __restrict__ e, const float* __restrict__ ce,
                       const float* __restrict__ wprop, const float* __restrict__ wsd,
                       const float* __restrict__ dmin, const float* __restrict__ dmax,
                       const float* __restrict__ cdmin, const float* __restrict__ cdmax,
                       unsigned short* __restrict__ Xb, unsigned short* __restrict__ Yb) {
  const int idx = blockIdx.x * blockDim.x + threadIdx.x;
  if (idx >= Bn*Cn*HWn) return;
  const int w = idx % Wn;
  const int h = (idx / Wn) % Hn;
  const int c = (idx / HWn) % Cn;
  const int b = idx / (HWn * Cn);
  const int bc = b*Cn + c;
  const int base = bc * Hp * Wp;
  const float wp = wprop[c];
  const int dy1[4] = {-1,-1,-1,0};
  const int dx1[4] = {-1, 0, 1,1};
  const int p = (b*Hn + h)*Wn + w;
  unsigned short xo[4], yo[4];
  #pragma unroll
  for (int k = 0; k < 4; ++k) {
    const int q1 = base + (h+1+dy1[k])*Wp + (w+1+dx1[k]);
    const int q2 = base + (h+1-dy1[k])*Wp + (w+1-dx1[k]);
    float dr0 = dmin[q1] / (dmax[q2] + EPSF);
    dr0 = __powf(fminf(fmaxf(dr0, EPSF), 1.0f), wsd[(c*2+0)*4 + k]);
    const float cr0 = cdmin[q1] * cdmax[q2];
    float dr1 = dmax[q1] / (dmin[q2] + EPSF);
    dr1 = __powf(fminf(fmaxf(dr1, EPSF), 1.0f), wsd[(c*2+1)*4 + k]);
    const float cr1 = cdmax[q1] * cdmin[q2];
    const bool sel1 = (cr1/dr1) > (cr0/dr0);   // argmax, first-wins
    const float efd  = sel1 ? dr1 : dr0;
    const float cefd = sel1 ? cr1 : cr0;
    const int ei = (bc*4 + k)*HWn + h*Wn + w;
    const float ev = e[ei], cev = ce[ei];
    const float den = wp*cev + cefd;
    const float e2  = (wp*cev*ev + cefd*efd) / (den + EPSF);
    const float ce2 = den / (wp + 1.0f);
    xo[k] = f2bf(ce2 * e2);
    yo[k] = f2bf(ce2);
  }
  uint2 xp, yp;
  xp.x = (unsigned)xo[0] | ((unsigned)xo[1] << 16);
  xp.y = (unsigned)xo[2] | ((unsigned)xo[3] << 16);
  yp.x = (unsigned)yo[0] | ((unsigned)yo[1] << 16);
  yp.y = (unsigned)yo[2] | ((unsigned)yo[3] << 16);
  *reinterpret_cast<uint2*>(Xb + (size_t)p*C4 + c*4) = xp;
  *reinterpret_cast<uint2*>(Yb + (size_t)p*C4 + c*4) = yp;
}

// ---------------------------------------------------------------- stage 2:
// dual GEMM (nom/den share B) with v_wmma_f32_16x16x32_bf16, LDS-transposed epilogue
__global__ void __launch_bounds__(256) k_gemm(const unsigned short* __restrict__ Xb,
                                              const unsigned short* __restrict__ Yb,
                                              const unsigned short* __restrict__ CWb,
                                              const float* __restrict__ sums,
                                              float* __restrict__ Af, float* __restrict__ Cf) {
  __shared__ float smem[2*32*128];              // 32 KB: [arr][ch 32][px 128]
  const int tid  = threadIdx.x;
  const int wave = tid >> 5;
  const int lane = tid & 31;
  const int mg = blockIdx.x;                    // 0..1151  (8 M-tiles per block)
  const int ng = blockIdx.y;                    // 0..3     (32 out-channels per block)
  const int prow = (mg*8 + wave)*16 + (lane & 15);
  const __bf16* X  = reinterpret_cast<const __bf16*>(Xb) + (size_t)prow*C4;
  const __bf16* Y  = reinterpret_cast<const __bf16*>(Yb) + (size_t)prow*C4;
  const __bf16* B0 = reinterpret_cast<const __bf16*>(CWb) + (size_t)(ng*32 + (lane & 15))*C4;
  const __bf16* B1 = B0 + 16*C4;
  const int hiA = (lane < 16) ? 0 : 8;          // D layout: lanes>=16 hold rows M+8
  const int hiB = (lane < 16) ? 0 : 16;         // contiguous k for A and B: lanes>=16 hold K+16..K+31
  v8f ax0 = {}, ax1 = {}, ay0 = {}, ay1 = {};
  #pragma unroll
  for (int ks = 0; ks < 4; ++ks) {
    const int k0 = ks*32;
    const v8bf xl = *reinterpret_cast<const v8bf*>(X + k0 + hiB);
    const v8bf xh = *reinterpret_cast<const v8bf*>(X + k0 + hiB + 8);
    const v8bf yl = *reinterpret_cast<const v8bf*>(Y + k0 + hiB);
    const v8bf yh = *reinterpret_cast<const v8bf*>(Y + k0 + hiB + 8);
    const v16bf avx = __builtin_shufflevector(xl, xh, 0,1,2,3,4,5,6,7,8,9,10,11,12,13,14,15);
    const v16bf avy = __builtin_shufflevector(yl, yh, 0,1,2,3,4,5,6,7,8,9,10,11,12,13,14,15);
    const v16bf w0 = *reinterpret_cast<const v16bf*>(B0 + k0 + hiB);
    const v16bf w1 = *reinterpret_cast<const v16bf*>(B1 + k0 + hiB);
    ax0 = __builtin_amdgcn_wmma_f32_16x16x32_bf16(false, avx, false, w0, (short)0, ax0, false, false);
    ax1 = __builtin_amdgcn_wmma_f32_16x16x32_bf16(false, avx, false, w1, (short)0, ax1, false, false);
    ay0 = __builtin_amdgcn_wmma_f32_16x16x32_bf16(false, avy, false, w0, (short)0, ay0, false, false);
    ay1 = __builtin_amdgcn_wmma_f32_16x16x32_bf16(false, avy, false, w1, (short)0, ay1, false, false);
  }
  const float sumCW = sums[0];
  #pragma unroll
  for (int nt = 0; nt < 2; ++nt) {
    const v8f nom = nt ? ax1 : ax0;
    const v8f den = nt ? ay1 : ay0;
    const int ch = nt*16 + (lane & 15);
    #pragma unroll
    for (int v = 0; v < 8; ++v) {               // D layout: M = v + (lane<16?0:8)
      const float dd  = den[v];
      const float e3  = nom[v] / (dd + EPSF);
      const float ce3 = dd / (sumCW + EPSF);
      const int px = wave*16 + v + hiA;
      smem[0*4096 + ch*128 + px] = ce3 * e3;
      smem[1*4096 + ch*128 + px] = ce3;
    }
  }
  __syncthreads();
  const int pbase = mg*128;
  #pragma unroll
  for (int i = 0; i < 32; ++i) {                // coalesced planar writes
    const int flat = i*256 + tid;
    const int arr = flat >> 12;
    const int rem = flat & 4095;
    const int ch  = rem >> 7;
    const int px  = rem & 127;
    float* dst = arr ? Cf : Af;
    dst[(size_t)(ng*32 + ch)*Pn + pbase + px] = smem[flat];
  }
}

// ---------------------------------------------------------------- stage 3:
// depthwise 5x5 (pad 2) on both maps + final normalization; inputs mostly L2-resident
__global__ void k_dw(const float* __restrict__ Af, const float* __restrict__ Cf,
                     const float* __restrict__ sw, const float* __restrict__ sums,
                     float* __restrict__ out) {
  const int idx = blockIdx.x * blockDim.x + threadIdx.x;
  if (idx >= Bn*C4*HWn) return;
  const int w = idx % Wn;
  const int h = (idx / Wn) % Hn;
  const int m = (idx / HWn) % C4;
  const int b = idx / (HWn * C4);
  const float* __restrict__ Ap = Af + (size_t)m*Pn + (size_t)b*HWn;
  const float* __restrict__ Cp = Cf + (size_t)m*Pn + (size_t)b*HWn;
  const float* __restrict__ swp = sw + m*25;
  float sA = 0.f, sC = 0.f;
  #pragma unroll
  for (int ky = 0; ky < 5; ++ky) {
    const int y = h + ky - 2;
    if (y < 0 || y >= Hn) continue;
    #pragma unroll
    for (int kx = 0; kx < 5; ++kx) {
      const int x = w + kx - 2;
      if (x < 0 || x >= Wn) continue;
      const float wgt = swp[ky*5 + kx];
      sA = fmaf(Ap[y*Wn + x], wgt, sA);
      sC = fmaf(Cp[y*Wn + x], wgt, sC);
    }
  }
  const float sumSW = sums[1];
  out[idx] = sA / (sC + EPSF);
  out[(size_t)Bn*C4*HWn + idx] = sC / (sumSW + EPSF);
}

// ----------------------------------------------------------------
extern "C" void kernel_launch(void* const* d_in, const int* in_sizes, int n_in,
                              void* d_out, int out_size, void* d_ws, size_t ws_size,
                              hipStream_t stream) {
  const float* d     = (const float*)d_in[0];
  const float* cd    = (const float*)d_in[1];
  const float* e     = (const float*)d_in[2];
  const float* ce    = (const float*)d_in[3];
  const float* wprop = (const float*)d_in[4];
  const float* wsd   = (const float*)d_in[5];
  const float* cw    = (const float*)d_in[6];
  const float* sw    = (const float*)d_in[7];
  float* out = (float*)d_out;

  char* ws = (char*)d_ws;
  size_t off = 0;
  auto alloc = [&](size_t bytes) -> char* {
    char* p = ws + off;
    off += (bytes + 255) & ~(size_t)255;
    return p;
  };
  float* dminW  = (float*)alloc((size_t)STATS_ELEMS * 4);
  float* dmaxW  = (float*)alloc((size_t)STATS_ELEMS * 4);
  float* cdminW = (float*)alloc((size_t)STATS_ELEMS * 4);
  float* cdmaxW = (float*)alloc((size_t)STATS_ELEMS * 4);
  unsigned short* Xb = (unsigned short*)alloc((size_t)Pn * C4 * 2);
  unsigned short* Yb = (unsigned short*)alloc((size_t)Pn * C4 * 2);
  float* Af = (float*)alloc((size_t)Pn * C4 * 4);
  float* Cf = (float*)alloc((size_t)Pn * C4 * 4);
  unsigned short* cwb = (unsigned short*)alloc((size_t)C4 * C4 * 2);
  float* sums = (float*)alloc(2 * 4);

  k_prep<<<1, 256, 0, stream>>>(cw, sw, cwb, sums);
  k_stats<<<(STATS_ELEMS + 255)/256, 256, 0, stream>>>(d, cd, dminW, dmaxW, cdminW, cdmaxW);
  k_fuse<<<(Bn*Cn*HWn + 255)/256, 256, 0, stream>>>(e, ce, wprop, wsd,
                                                    dminW, dmaxW, cdminW, cdmaxW, Xb, Yb);
  dim3 g2(Pn/128, 4);   // 1152 x 4 blocks, 256 threads (8 waves) each
  k_gemm<<<g2, 256, 0, stream>>>(Xb, Yb, cwb, sums, Af, Cf);
  k_dw<<<(Bn*C4*HWn + 255)/256, 256, 0, stream>>>(Af, Cf, sw, sums, out);
}
